// PredHead_82308753260988
// MI455X (gfx1250) — matrix operand order repaired and checked by reading, not compiled, not run on hardware
//
#include <hip/hip_runtime.h>
#include <math.h>

#define B_  4
#define L_  384
#define DS_ 384
#define DP_ 128

typedef __attribute__((ext_vector_type(16))) __bf16 v16bf_t;
typedef __attribute__((ext_vector_type(8)))  float  v8f_t;

union BFrag { v16bf_t v; unsigned short u[16]; };

__device__ __forceinline__ unsigned short f2bf(float f) {
  unsigned int u = __float_as_uint(f);
  u += 0x7FFFu + ((u >> 16) & 1u);        // round-to-nearest-even
  return (unsigned short)(u >> 16);
}

__device__ __forceinline__ float sigmoidf_(float x) { return 1.f / (1.f + expf(-x)); }

// ---------------------------------------------------------------------------
// zero scratch
// ---------------------------------------------------------------------------
__global__ void k_zero(float* __restrict__ p, int n) {
  int t = blockIdx.x * blockDim.x + threadIdx.x;
  if (t < n) p[t] = 0.f;
}

// ---------------------------------------------------------------------------
// Dominant kernel: stream pair_repr once per (row, transposed-column), apply
// symmetrize + zero-diag + LayerNorm(128) in registers, and accumulate the
// masked j-reductions.  One workgroup (8 waves) per (b,i); each lane owns 4
// channels (32 lanes * 4 = 128). LN stats via wave32 shfl_xor reduction.
// ---------------------------------------------------------------------------
__global__ __launch_bounds__(256)
void k_pair_reduce(const float* __restrict__ P,       // [B,L,L,128]
                   const float* __restrict__ gz,      // [128]
                   const float* __restrict__ bz,      // [128]
                   const int*   __restrict__ chain,   // [B,L]
                   const unsigned char* __restrict__ mask, // [B,L]
                   float* __restrict__ rowsum,        // [B*L,128]  sum_j LN * inter
                   float* __restrict__ rowcnt,        // [B*L]
                   float* __restrict__ bandsum,       // [B,128]    (atomic)
                   float* __restrict__ bandcnt) {     // [B]        (atomic)
  const int bi   = blockIdx.x;
  const int b    = bi / L_;
  const int i    = bi % L_;
  const int lane = threadIdx.x & 31;
  const int wave = threadIdx.x >> 5;

  const int mi = mask[b * L_ + i];
  const int ci = chain[b * L_ + i];

  const float4 g  = ((const float4*)gz)[lane];
  const float4 bb = ((const float4*)bz)[lane];

  float4 accI = {0.f, 0.f, 0.f, 0.f};
  float4 accB = {0.f, 0.f, 0.f, 0.f};
  float  cntI = 0.f, cntB = 0.f;

  const float* baseRow = P + (size_t)(b * L_ + i) * (L_ * DP_);
  const float* baseCol = P + (size_t)b * L_ * L_ * DP_ + (size_t)i * DP_;

  if (mi) {
    for (int j = wave; j < L_; j += 8) {           // j uniform across the wave
      if (!mask[b * L_ + j]) continue;
      const bool same = (chain[b * L_ + j] == ci);
      int dij = i - j; if (dij < 0) dij = -dij;
      const bool isInter = !same;
      const bool isBand  = same && (dij <= 4);
      if (!isInter && !isBand) continue;           // skip the loads entirely

      const float4 x = *(const float4*)(baseRow + (size_t)j * DP_ + lane * 4);
      const float4 y = *(const float4*)(baseCol + (size_t)j * (L_ * DP_) + lane * 4);
      const float ds = (i == j) ? 0.f : 0.5f;      // zero diagonal
      float4 s;
      s.x = ds * (x.x + y.x); s.y = ds * (x.y + y.y);
      s.z = ds * (x.z + y.z); s.w = ds * (x.w + y.w);

      float sum = s.x + s.y + s.z + s.w;
      float ssq = s.x * s.x + s.y * s.y + s.z * s.z + s.w * s.w;
      #pragma unroll
      for (int off = 16; off > 0; off >>= 1) {
        sum += __shfl_xor(sum, off, 32);
        ssq += __shfl_xor(ssq, off, 32);
      }
      const float mu  = sum * (1.f / 128.f);
      const float var = ssq * (1.f / 128.f) - mu * mu;
      const float rs  = rsqrtf(var + 1e-5f);       // LN(0-vec) -> bias, as in ref
      float4 v;
      v.x = (s.x - mu) * rs * g.x + bb.x;
      v.y = (s.y - mu) * rs * g.y + bb.y;
      v.z = (s.z - mu) * rs * g.z + bb.z;
      v.w = (s.w - mu) * rs * g.w + bb.w;

      if (isInter) {
        accI.x += v.x; accI.y += v.y; accI.z += v.z; accI.w += v.w; cntI += 1.f;
      } else {
        accB.x += v.x; accB.y += v.y; accB.z += v.z; accB.w += v.w; cntB += 1.f;
      }
    }
  }

  __shared__ float ldsI[8 * 128];
  __shared__ float ldsB[8 * 128];
  __shared__ float ldsCI[8], ldsCB[8];
  ((float4*)(ldsI + wave * 128))[lane] = accI;
  ((float4*)(ldsB + wave * 128))[lane] = accB;
  if (lane == 0) { ldsCI[wave] = cntI; ldsCB[wave] = cntB; }
  __syncthreads();

  const int tid = threadIdx.x;
  if (tid < 128) {
    float sI = 0.f, sB = 0.f;
    #pragma unroll
    for (int w = 0; w < 8; ++w) { sI += ldsI[w * 128 + tid]; sB += ldsB[w * 128 + tid]; }
    rowsum[(size_t)bi * 128 + tid] = sI;
    if (sB != 0.f) atomicAdd(&bandsum[b * 128 + tid], sB);
  } else if (tid == 128) {
    float cI = 0.f, cB = 0.f;
    #pragma unroll
    for (int w = 0; w < 8; ++w) { cI += ldsCI[w]; cB += ldsCB[w]; }
    rowcnt[bi] = cI;
    if (cB != 0.f) atomicAdd(&bandcnt[b], cB);
  }
}

// ---------------------------------------------------------------------------
// LayerNorm of seq_repr rows (384 channels). One wave per row, 12 ch/lane.
// ---------------------------------------------------------------------------
__global__ __launch_bounds__(256)
void k_seq_ln(const float* __restrict__ X, const float* __restrict__ g,
              const float* __restrict__ bb, float* __restrict__ Y) {
  const int lane = threadIdx.x & 31;
  const int wave = threadIdx.x >> 5;
  const int row  = blockIdx.x * 8 + wave;          // 0..1535
  const float* x = X + (size_t)row * DS_;
  float v[12];
  float sum = 0.f, ssq = 0.f;
  #pragma unroll
  for (int t = 0; t < 12; ++t) {
    v[t] = x[lane + 32 * t];
    sum += v[t]; ssq += v[t] * v[t];
  }
  #pragma unroll
  for (int off = 16; off > 0; off >>= 1) {
    sum += __shfl_xor(sum, off, 32);
    ssq += __shfl_xor(ssq, off, 32);
  }
  const float mu  = sum * (1.f / 384.f);
  const float var = ssq * (1.f / 384.f) - mu * mu;
  const float rs  = rsqrtf(var + 1e-5f);
  float* y = Y + (size_t)row * DS_;
  #pragma unroll
  for (int t = 0; t < 12; ++t) {
    const int c = lane + 32 * t;
    y[c] = (v[t] - mu) * rs * g[c] + bb[c];
  }
}

// ---------------------------------------------------------------------------
// bf16 WMMA GEMM: C[M,N] = A[M,K] * Bw[K,N], row-major, f32 in/out with
// inline RNE conversion into v_wmma_f32_16x16x32_bf16 fragments.
// One wave per 16x16 output tile (EXEC all-ones, no divergence).
// A/B fragments use contiguous k (k = 16*half + t) in both operands (wave32, 16-bit A 16x32 / B 32x16).
// ---------------------------------------------------------------------------
__global__ __launch_bounds__(32)
void k_gemm_bf16(const float* __restrict__ A, const float* __restrict__ Bw,
                 float* __restrict__ C, int M, int N, int K) {
  const int lane = threadIdx.x;
  const int half = lane >> 4;                      // 0 or 1
  const int l16  = lane & 15;
  const int row0 = blockIdx.y << 4;
  const int col0 = blockIdx.x << 4;
  const float* Arow = A + (size_t)(row0 + l16) * K;
  const int bcol = col0 + l16;

  v8f_t acc = {0.f, 0.f, 0.f, 0.f, 0.f, 0.f, 0.f, 0.f};

  for (int kb = 0; kb < K; kb += 32) {
    BFrag af, bf;
    // A 16x32: lanes 0-15 row M=l16 hold K = kb..kb+15;
    //          lanes 16-31 hold K = kb+16..kb+31
    #pragma unroll
    for (int t = 0; t < 16; ++t) {
      af.u[t] = f2bf(Arow[kb + half * 16 + t]);
    }
    // B 32x16: lane holds column N=l16; lanes 0-15 -> K=kb..kb+15,
    //          lanes 16-31 -> K=kb+16..kb+31
    #pragma unroll
    for (int t = 0; t < 16; ++t) {
      bf.u[t] = f2bf(Bw[(size_t)(kb + half * 16 + t) * N + bcol]);
    }
    acc = __builtin_amdgcn_wmma_f32_16x16x32_bf16(false, af.v, false, bf.v,
                                                  (short)0, acc, false, false);
  }
  // C/D 16x16 f32: VGPR r -> M = r + 8*half, N = l16
  #pragma unroll
  for (int r = 0; r < 8; ++r) {
    C[(size_t)(row0 + half * 8 + r) * N + (col0 + l16)] = acc[r];
  }
}

// ---------------------------------------------------------------------------
// Gating + MLP head. One 256-thread block per batch; everything is tiny.
// ---------------------------------------------------------------------------
__global__ __launch_bounds__(256)
void k_head(const float* __restrict__ S,        // [B,L,384]  s after W_seq
            const float* __restrict__ rowmat,   // [B,L,128]  rowsum @ W_pair
            const float* __restrict__ rowcnt,   // [B,L]
            const float* __restrict__ bandsum,  // [B,128]
            const float* __restrict__ bandcnt,  // [B]
            const unsigned char* __restrict__ mask,
            const float* __restrict__ Wgs, const float* __restrict__ bgs,
            const float* __restrict__ Wgz, const float* __restrict__ bgz,
            const float* __restrict__ Wpair,
            const float* __restrict__ W1, const float* __restrict__ b1,
            const float* __restrict__ W2, const float* __restrict__ b2,
            float* __restrict__ out) {
  const int b   = blockIdx.x;
  const int tid = threadIdx.x;

  __shared__ float smean[DS_], smax[DS_];
  __shared__ float imean[DP_], ipeak[DP_];
  __shared__ float feat[DS_ + 2 * DP_];
  __shared__ float rc[L_];
  __shared__ float bsum[DP_];
  __shared__ float hbuf[64];
  __shared__ float denm, denI;

  // phase 0: stage small vectors, compute denominators
  for (int l = tid; l < L_; l += 256) rc[l] = rowcnt[b * L_ + l];
  if (tid < DP_) bsum[tid] = bandsum[b * DP_ + tid];
  if (tid == 0) {
    float d = 0.f;
    for (int l = 0; l < L_; ++l) d += mask[b * L_ + l] ? 1.f : 0.f;
    denm = d > 1e-6f ? d : 1e-6f;
  }
  if (tid == 1) {
    float d = 0.f;
    for (int l = 0; l < L_; ++l) d += rowcnt[b * L_ + l];
    denI = d > 1e-6f ? d : 1e-6f;
  }
  __syncthreads();

  // phase 1: sequence masked mean/max, inter mean/peak
  for (int c = tid; c < DS_; c += 256) {
    float sm = 0.f, mx = -3.0e38f; bool any = false;
    for (int l = 0; l < L_; ++l) {
      if (mask[b * L_ + l]) {
        const float v = S[(size_t)(b * L_ + l) * DS_ + c];
        sm += v; if (v > mx) mx = v; any = true;
      }
    }
    smean[c] = sm / denm;
    smax[c]  = any ? mx : 0.f;
  }
  if (tid < DP_) {
    const int c = tid;
    float s = 0.f, pk = -3.0e38f;
    for (int l = 0; l < L_; ++l) {
      const float v = rowmat[(size_t)(b * L_ + l) * DP_ + c];
      s += v;
      const float d = rc[l] > 1e-6f ? rc[l] : 1e-6f;
      const float m = v / d;                        // row_mean (all rows, like ref)
      if (m > pk) pk = m;
    }
    imean[c] = s / denI;
    ipeak[c] = pk;
  }
  __syncthreads();

  // phase 2: gates + feature assembly
  for (int c = tid; c < DS_; c += 256) {
    float acc = bgs[c];
    for (int k = 0; k < DS_; ++k) acc += smax[k]  * Wgs[k * DS_ + c];
    for (int k = 0; k < DS_; ++k) acc += smean[k] * Wgs[(DS_ + k) * DS_ + c];
    const float gs = sigmoidf_(acc);
    feat[c] = smean[c] + gs * (smax[c] - smean[c]);
  }
  if (tid < DP_) {
    const int c = tid;
    float acc = bgz[c];
    for (int k = 0; k < DP_; ++k) acc += ipeak[k] * Wgz[k * DP_ + c];
    for (int k = 0; k < DP_; ++k) acc += imean[k] * Wgz[(DP_ + k) * DP_ + c];
    const float gzv = sigmoidf_(acc);
    feat[DS_ + c] = imean[c] + gzv * (ipeak[c] - imean[c]);
    // intra_compact = (bandsum @ W_pair) / bandcnt   (matmul after reduction)
    float bv = 0.f;
    for (int k = 0; k < DP_; ++k) bv += bsum[k] * Wpair[k * DP_ + c];
    const float dB = bandcnt[b] > 1e-6f ? bandcnt[b] : 1e-6f;
    feat[DS_ + DP_ + c] = bv / dB;
  }
  __syncthreads();

  // phase 3: MLP head
  if (tid < 64) {
    float acc = b1[tid];
    for (int k = 0; k < DS_ + 2 * DP_; ++k) acc += feat[k] * W1[k * 64 + tid];
    hbuf[tid] = acc > 0.f ? acc : 0.f;
  }
  __syncthreads();
  if (tid == 0) {
    float logit = b2[0];
    for (int k = 0; k < 64; ++k) logit += hbuf[k] * W2[k];
    out[b] = sigmoidf_(logit);
  }
}

// ---------------------------------------------------------------------------
extern "C" void kernel_launch(void* const* d_in, const int* in_sizes, int n_in,
                              void* d_out, int out_size, void* d_ws, size_t ws_size,
                              hipStream_t stream) {
  (void)in_sizes; (void)n_in; (void)out_size; (void)ws_size;

  const float* seq   = (const float*)d_in[0];
  const float* pair  = (const float*)d_in[1];
  const float* ln_sg = (const float*)d_in[2];
  const float* ln_sb = (const float*)d_in[3];
  const float* ln_zg = (const float*)d_in[4];
  const float* ln_zb = (const float*)d_in[5];
  const float* Wseq  = (const float*)d_in[6];
  const float* Wpair = (const float*)d_in[7];
  const float* Wgs   = (const float*)d_in[8];
  const float* bgs   = (const float*)d_in[9];
  const float* Wgz   = (const float*)d_in[10];
  const float* bgz   = (const float*)d_in[11];
  const float* W1    = (const float*)d_in[12];
  const float* b1    = (const float*)d_in[13];
  const float* W2    = (const float*)d_in[14];
  const float* b2    = (const float*)d_in[15];
  const int*   chain = (const int*)d_in[16];
  const unsigned char* mask = (const unsigned char*)d_in[17];
  float* out = (float*)d_out;

  char* ws = (char*)d_ws;
  size_t off = 0;
  auto carve = [&](size_t bytes) -> float* {
    float* p = (float*)(ws + off);
    off += (bytes + 255) & ~(size_t)255;
    return p;
  };
  float* rowsum  = carve((size_t)B_ * L_ * DP_ * 4);   // 786 KB
  float* rowcnt  = carve((size_t)B_ * L_ * 4);
  float* bandblk = carve((size_t)(B_ * DP_ + B_) * 4);
  float* bandsum = bandblk;
  float* bandcnt = bandblk + B_ * DP_;
  float* sln     = carve((size_t)B_ * L_ * DS_ * 4);   // 2.36 MB
  float* sout    = carve((size_t)B_ * L_ * DS_ * 4);   // 2.36 MB
  float* rowmat  = carve((size_t)B_ * L_ * DP_ * 4);   // 786 KB

  k_zero<<<3, 256, 0, stream>>>(bandblk, B_ * DP_ + B_);

  // dominant: stream pair_repr, LN + masked j-reductions
  k_pair_reduce<<<B_ * L_, 256, 0, stream>>>(pair, ln_zg, ln_zb, chain, mask,
                                             rowsum, rowcnt, bandsum, bandcnt);

  // sequence branch LN
  k_seq_ln<<<(B_ * L_) / 8, 256, 0, stream>>>(seq, ln_sg, ln_sb, sln);

  // WMMA GEMMs: sln @ W_seq  and  rowsum @ W_pair
  k_gemm_bf16<<<dim3(DS_ / 16, (B_ * L_) / 16), 32, 0, stream>>>(
      sln, Wseq, sout, B_ * L_, DS_, DS_);
  k_gemm_bf16<<<dim3(DP_ / 16, (B_ * L_) / 16), 32, 0, stream>>>(
      rowsum, Wpair, rowmat, B_ * L_, DP_, DP_);

  // gating + MLP head
  k_head<<<B_, 256, 0, stream>>>(sout, rowmat, rowcnt, bandsum, bandcnt, mask,
                                 Wgs, bgs, Wgz, bgz, Wpair, W1, b1, W2, b2, out);
}
